// LocalEnhanceAttention_88295937671568
// MI455X (gfx1250) — matrix operand order repaired and checked by reading, not compiled, not run on hardware
//
#include <hip/hip_runtime.h>

// LocalEnhanceAttention (window cosine attention) for MI455X / gfx1250.
// B=32, C=256 (8 heads x d=32), H=W=56, window 7x7 (49 tokens, padded to 64).
// One workgroup (4 x wave32) per (batch, window, head) tile; 16384 workgroups.
// QK^T and P*V via v_wmma_f32_16x16x32_f16; softmax in f32 registers.

#define B_   32
#define C_   256
#define H_   56
#define W_   56
#define NH   8
#define DH   32    // head dim
#define WSZ  7
#define S_   49    // tokens per window
#define SP   64    // padded token count
#define CLAMP_MAX 4.6052f

typedef _Float16 v16h __attribute__((ext_vector_type(16)));
typedef _Float16 v8h  __attribute__((ext_vector_type(8)));
typedef float    v8f  __attribute__((ext_vector_type(8)));

// A-operand (16x32 f16, MxK): as loaded here, lane<16 holds K=0..15 (same order as B),
// lane>=16 holds K=16..31 for its row. Both chunks are 16B-contiguous.
__device__ __forceinline__ v16h make_a(const _Float16* p0, const _Float16* p1) {
  v8h lo = *(const v8h*)p0;
  v8h hi = *(const v8h*)p1;
  v16h a;
#pragma unroll
  for (int i = 0; i < 8; ++i) { a[i] = lo[i]; a[i + 8] = hi[i]; }
  return a;
}

// B-operand (32x16 f16, KxN): lane n holds column n&15; lanes<16 K=0..15,
// lanes>=16 K=16..31 -> one 32B-contiguous run.
__device__ __forceinline__ v16h make_b(const _Float16* p) {
  v8h lo = *(const v8h*)p;
  v8h hi = *(const v8h*)(p + 8);
  v16h b;
#pragma unroll
  for (int i = 0; i < 8; ++i) { b[i] = lo[i]; b[i + 8] = hi[i]; }
  return b;
}

__global__ __launch_bounds__(128)
void lea_win_attn_kernel(const float* __restrict__ q,
                         const float* __restrict__ k,
                         const float* __restrict__ v,
                         const float* __restrict__ logit_scale,
                         float* __restrict__ out) {
  __shared__ __align__(16) float    qf[S_ * DH];   // raw f32 Q (for L2 norm)
  __shared__ __align__(16) float    kf[S_ * DH];   // raw f32 K
  __shared__            float       rq[S_];
  __shared__            float       rk[S_];
  __shared__ __align__(16) _Float16 Qh[SP * DH];   // normalized f16, row-major, zero-padded
  __shared__ __align__(16) _Float16 Kh[SP * DH];
  __shared__ __align__(16) _Float16 Vt[DH * SP];   // V transposed: [dd][t], zero-padded
  __shared__ __align__(16) _Float16 Pm[SP * SP];   // softmax probs, row-major
  __shared__ __align__(16) float    Of[SP * DH];   // f32 output tile

  const int tid  = threadIdx.x;
  const int lane = tid & 31;
  const int wv   = tid >> 5;        // wave id 0..3 -> M-stripe rows [16*wv, 16*wv+15]
  const int hrow = lane & 15;
  const int hsel = lane >> 4;       // 0: lanes 0-15, 1: lanes 16-31

  int idx = blockIdx.x;
  const int wx = idx & 7;  idx >>= 3;
  const int wy = idx & 7;  idx >>= 3;
  const int hd = idx & 7;  idx >>= 3;
  const int b  = idx;

  const float scale = __expf(fminf(logit_scale[hd], CLAMP_MAX));

  const size_t plane = (size_t)H_ * W_;
  const size_t base  = (((size_t)b * C_ + hd * DH) * H_ + wy * WSZ) * (size_t)W_ + wx * WSZ;

  // --- zero padded operand tiles ---
  for (int i = tid; i < SP * DH; i += 128) {
    Qh[i] = (_Float16)0.f; Kh[i] = (_Float16)0.f; Vt[i] = (_Float16)0.f;
  }
  __syncthreads();

  // --- gather q/k/v for this window (7-float contiguous runs per (dd,ty)) ---
  for (int i = tid; i < S_ * DH; i += 128) {
    const int dd = i / S_;
    const int t  = i - dd * S_;
    const int ty = t / WSZ, tx = t - ty * WSZ;
    const size_t g = base + (size_t)dd * plane + (size_t)ty * W_ + tx;
    qf[t * DH + dd] = q[g];
    kf[t * DH + dd] = k[g];
    Vt[dd * SP + t] = (_Float16)v[g];   // transposed store: B-fragments become contiguous
  }
  __syncthreads();

  // --- L2 row norms (x / max(||x||, 1e-12)) ---
  if (tid < S_) {
    float s = 0.f;
#pragma unroll
    for (int d = 0; d < DH; ++d) { const float x = qf[tid * DH + d]; s += x * x; }
    rq[tid] = 1.0f / fmaxf(sqrtf(s), 1e-12f);
  } else if (tid >= 64 && tid < 64 + S_) {
    const int t = tid - 64;
    float s = 0.f;
#pragma unroll
    for (int d = 0; d < DH; ++d) { const float x = kf[t * DH + d]; s += x * x; }
    rk[t] = 1.0f / fmaxf(sqrtf(s), 1e-12f);
  }
  __syncthreads();

  // --- normalize + convert to f16 operands ---
  for (int i = tid; i < S_ * DH; i += 128) {
    const int t = i >> 5;            // DH == 32
    Qh[i] = (_Float16)(qf[i] * rq[t]);
    Kh[i] = (_Float16)(kf[i] * rk[t]);
  }
  __syncthreads();

  // --- dots = Qn * Kn^T for this wave's 16-row stripe (K-dim = 32, one step) ---
  v8f acc[4];
  {
    const _Float16* qp = &Qh[(wv * 16 + hrow) * DH + hsel * 16];
    const v16h aq = make_a(qp, qp + 8);
#pragma unroll
    for (int j = 0; j < 4; ++j) {
      const _Float16* kp = &Kh[(j * 16 + hrow) * DH + hsel * 16];
      const v16h bk = make_b(kp);
      v8f z = {};
      acc[j] = __builtin_amdgcn_wmma_f32_16x16x32_f16(
          false, aq, false, bk, (short)0, z, false, false);
    }
  }

  // --- scaled, masked softmax over t (N dimension), per accumulator row slot ---
#pragma unroll
  for (int r = 0; r < 8; ++r) {
    float vals[4];
    float m = -3.0e38f;
#pragma unroll
    for (int j = 0; j < 4; ++j) {
      const int col = j * 16 + hrow;
      float x = acc[j][r] * scale;
      x = (col < S_) ? x : -3.0e38f;   // mask padded keys
      vals[j] = x;
      m = fmaxf(m, x);
    }
#pragma unroll
    for (int off = 1; off < 16; off <<= 1) m = fmaxf(m, __shfl_xor(m, off, 32));
    float ssum = 0.f;
#pragma unroll
    for (int j = 0; j < 4; ++j) { const float e = __expf(vals[j] - m); vals[j] = e; ssum += e; }
#pragma unroll
    for (int off = 1; off < 16; off <<= 1) ssum += __shfl_xor(ssum, off, 32);
    const float inv = 1.0f / ssum;
    const int row = wv * 16 + r + hsel * 8;
#pragma unroll
    for (int j = 0; j < 4; ++j)
      Pm[row * SP + j * 16 + hrow] = (_Float16)(vals[j] * inv);
  }
  __syncthreads();

  // --- out = P * V  (M=16 stripe, N=32 in two tiles, K=64 in two steps) ---
  v8f o[2] = {};
#pragma unroll
  for (int kk = 0; kk < 2; ++kk) {
    const _Float16* pp = &Pm[(wv * 16 + hrow) * SP + kk * 32 + hsel * 16];
    const v16h ap = make_a(pp, pp + 8);
#pragma unroll
    for (int j = 0; j < 2; ++j) {
      const _Float16* vp = &Vt[(j * 16 + hrow) * SP + kk * 32 + hsel * 16];
      const v16h bv = make_b(vp);
      o[j] = __builtin_amdgcn_wmma_f32_16x16x32_f16(
          false, ap, false, bv, (short)0, o[j], false, false);
    }
  }

  // --- park output tile in LDS for coalesced scatter ---
#pragma unroll
  for (int r = 0; r < 8; ++r) {
    const int row = wv * 16 + r + hsel * 8;
#pragma unroll
    for (int j = 0; j < 2; ++j)
      Of[row * DH + j * 16 + hrow] = o[j][r];
  }
  __syncthreads();

  // --- writeback (same gather pattern as loads) ---
  for (int i = tid; i < S_ * DH; i += 128) {
    const int dd = i / S_;
    const int t  = i - dd * S_;
    const int ty = t / WSZ, tx = t - ty * WSZ;
    out[base + (size_t)dd * plane + (size_t)ty * W_ + tx] = Of[t * DH + dd];
  }
}

extern "C" void kernel_launch(void* const* d_in, const int* in_sizes, int n_in,
                              void* d_out, int out_size, void* d_ws, size_t ws_size,
                              hipStream_t stream) {
  (void)in_sizes; (void)n_in; (void)out_size; (void)d_ws; (void)ws_size;
  const float* q  = (const float*)d_in[0];
  const float* k  = (const float*)d_in[1];
  const float* v  = (const float*)d_in[2];
  const float* ls = (const float*)d_in[3];
  float* out = (float*)d_out;

  const int n_tiles = B_ * 8 * 8 * NH;   // 16384 (batch, window, head) tiles
  lea_win_attn_kernel<<<dim3(n_tiles), dim3(128), 0, stream>>>(q, k, v, ls, out);
}
